// TopKMixtureOfExpertsBlock_80384607911983
// MI455X (gfx1250) — matrix operand order repaired and checked by reading, not compiled, not run on hardware
//
#include <hip/hip_runtime.h>
#include <hip/hip_bf16.h>

// ---------------------------------------------------------------------------
// MoE top-2 of 8, D=256, bf16 WMMA (v_wmma_f32_16x16x32_bf16) on gfx1250,
// with async global->LDS staging of the gathered token tile (ASYNCcnt path).
// ---------------------------------------------------------------------------

#define D_DIM 256
#define NEXP 8
#define TM 48          // gathered rows per block (3 x 16-row WMMA tiles)
#define TMH 24         // half-tile rows staged per async pass
#define LDH 264        // LDS row stride in ushorts (16B-aligned chunks, no pow2 conflicts)

typedef __attribute__((ext_vector_type(16))) __bf16 v16bf;
typedef __attribute__((ext_vector_type(8)))  float  v8f;
typedef __attribute__((ext_vector_type(8)))  unsigned short usvec8;
typedef __attribute__((ext_vector_type(4)))  unsigned short usvec4;
typedef __attribute__((ext_vector_type(4)))  int    v4i;

#if defined(__HIP_DEVICE_COMPILE__) && __has_builtin(__builtin_amdgcn_global_load_async_to_lds_b128)
#define HAS_ASYNC_LDS 1
typedef __attribute__((address_space(1))) void glb_void;
typedef __attribute__((address_space(3))) void lds_void;
typedef __attribute__((address_space(1))) v4i* glb_v4i_ptr;
typedef __attribute__((address_space(3))) v4i* lds_v4i_ptr;
#else
#define HAS_ASYNC_LDS 0
#endif

union FragBF { v16bf v; usvec8 h[2]; };

static __device__ inline unsigned short f32_to_bf16(float f) {
    unsigned int u = __builtin_bit_cast(unsigned int, f);
    u += 0x7FFFu + ((u >> 16) & 1u);   // round-to-nearest-even
    return (unsigned short)(u >> 16);
}

// A fragment: 16x32 bf16 from LDS (row-major, stride LDH). Per ISA 16-bit A layout:
// lanes 0-15 rows 0-15 with K chunks [k0..k0+7],[k0+16..k0+23]; lanes 16-31 shifted +8.
static __device__ inline FragBF load_frag_a(const unsigned short* sh, int row_base,
                                            int k0, int lane) {
    int m16 = lane & 15, hi = lane >> 4;
    const unsigned short* p = sh + (row_base + m16) * LDH + k0 + hi * 8;
    FragBF f;
    f.h[0] = *(const usvec8*)p;
    f.h[1] = *(const usvec8*)(p + 16);
    return f;
}

// B fragment: 32x16 bf16 from TRANSPOSED weights wt[n][k] (stride 256):
// lane n=lane&15, K chunks [k0+h*8..+7],[k0+16+h*8..+7] with h=lane>>4 (matches A layout).
static __device__ inline FragBF load_frag_b(const unsigned short* wt, int ncol_base,
                                            int k0, int lane) {
    int n  = ncol_base + (lane & 15);
    int kb = k0 + (lane >> 4) * 8;
    const unsigned short* p = wt + n * D_DIM + kb;
    FragBF f;
    f.h[0] = *(const usvec8*)p;
    f.h[1] = *(const usvec8*)(p + 16);
    return f;
}

static __device__ inline v8f wmma_bf16(const FragBF& a, const FragBF& b, v8f c) {
    return __builtin_amdgcn_wmma_f32_16x16x32_bf16(
        /*neg_a=*/false, a.v, /*neg_b=*/false, b.v,
        /*c_mod=*/(short)0, c, /*reuse_a=*/false, /*reuse_b=*/false);
}

#if HAS_ASYNC_LDS
static __device__ inline void wait_async0() {
#if __has_builtin(__builtin_amdgcn_s_wait_asynccnt)
    __builtin_amdgcn_s_wait_asynccnt(0);
#else
    asm volatile("s_wait_asynccnt 0" ::: "memory");
#endif
}
#endif

// ---------------------------------------------------------------------------
// 1) init: zero counters, invalidate gather slots
// ---------------------------------------------------------------------------
__global__ void moe_init_kernel(int* counts, int* counts2,
                                int* row_token, float* row_w, int M) {
    int i = blockIdx.x * 256 + threadIdx.x;
    if (i < NEXP) counts[i] = 0;
    if (i >= NEXP && i < 2 * NEXP) counts2[i - NEXP] = 0;
    if (i < M) { row_token[i] = -1; row_w[i] = 0.0f; }
}

// ---------------------------------------------------------------------------
// 2) router: wave-per-token logits -> softmax -> top2 -> weights + counts
// ---------------------------------------------------------------------------
__global__ __launch_bounds__(256) void moe_router_kernel(
    const float* __restrict__ tokens, const float* __restrict__ rw,
    int* __restrict__ r_e, float* __restrict__ r_w, int* __restrict__ counts, int N) {
    __shared__ float swr[NEXP * D_DIM];
    int tid = threadIdx.x;
    for (int i = tid; i < NEXP * D_DIM; i += 256) swr[i] = rw[i];
    __syncthreads();

    int wv = tid >> 5, lane = tid & 31;
    int t = blockIdx.x * 8 + wv;
    if (t >= N) return;

    const float* x = tokens + (size_t)t * D_DIM;
    float4 xa = *(const float4*)(x + lane * 8);
    float4 xb = *(const float4*)(x + lane * 8 + 4);
    float part[NEXP];
#pragma unroll
    for (int e = 0; e < NEXP; ++e) {
        const float* w = swr + e * D_DIM + lane * 8;
        float4 wa = *(const float4*)w;
        float4 wb = *(const float4*)(w + 4);
        part[e] = xa.x * wa.x + xa.y * wa.y + xa.z * wa.z + xa.w * wa.w +
                  xb.x * wb.x + xb.y * wb.y + xb.z * wb.z + xb.w * wb.w;
    }
#pragma unroll
    for (int off = 16; off >= 1; off >>= 1)
#pragma unroll
        for (int e = 0; e < NEXP; ++e)
            part[e] += __shfl_xor(part[e], off, 32);

    if (lane == 0) {
        float mx = part[0];
#pragma unroll
        for (int e = 1; e < NEXP; ++e) mx = fmaxf(mx, part[e]);
        float p[NEXP], s = 0.0f;
#pragma unroll
        for (int e = 0; e < NEXP; ++e) { p[e] = __expf(part[e] - mx); s += p[e]; }
        float inv = 1.0f / s;
#pragma unroll
        for (int e = 0; e < NEXP; ++e) p[e] *= inv;
        int i1 = 0;
#pragma unroll
        for (int e = 1; e < NEXP; ++e) if (p[e] > p[i1]) i1 = e;
        int i2 = (i1 == 0) ? 1 : 0;
#pragma unroll
        for (int e = 0; e < NEXP; ++e) if (e != i1 && p[e] > p[i2]) i2 = e;
        float den = fmaxf(p[i1] + p[i2], 1e-12f);
        r_e[2 * t]     = i1;  r_e[2 * t + 1] = i2;
        r_w[2 * t]     = p[i1] / den;
        r_w[2 * t + 1] = p[i2] / den;
        atomicAdd(&counts[i1], 1);
        atomicAdd(&counts[i2], 1);
    }
}

// ---------------------------------------------------------------------------
// 3) prep: fp32 [e][k][n] weights -> transposed bf16 [e][n][k]
// ---------------------------------------------------------------------------
__global__ void moe_prep_kernel(const float* __restrict__ w1, const float* __restrict__ w2,
                                unsigned short* __restrict__ w1t,
                                unsigned short* __restrict__ w2t) {
    int i = blockIdx.x * 256 + threadIdx.x;   // i < 8*256*256
    int e = i >> 16, rem = i & 65535, k = rem >> 8, n = rem & 255;
    int dst = (e << 16) | (n << 8) | k;
    w1t[dst] = f32_to_bf16(w1[i]);
    w2t[dst] = f32_to_bf16(w2[i]);
}

// ---------------------------------------------------------------------------
// 4) scan: TM-aligned per-expert segment bases (single thread; 8 experts)
// ---------------------------------------------------------------------------
__global__ void moe_scan_kernel(const int* __restrict__ counts, int* __restrict__ base) {
    if (threadIdx.x == 0 && blockIdx.x == 0) {
        int acc = 0;
        for (int e = 0; e < NEXP; ++e) {
            base[e] = acc;
            acc += ((counts[e] + TM - 1) / TM) * TM;
        }
        base[NEXP] = acc;
    }
}

// ---------------------------------------------------------------------------
// 5) gather: slot = base[e] + running position; record token id + weight
// ---------------------------------------------------------------------------
__global__ void moe_gather_kernel(const int* __restrict__ r_e, const float* __restrict__ r_w,
                                  const int* __restrict__ base, int* __restrict__ counts2,
                                  int* __restrict__ row_token, float* __restrict__ row_w,
                                  int M2) {
    int i = blockIdx.x * 256 + threadIdx.x;
    if (i >= M2) return;
    int e = r_e[i];
    int pos = atomicAdd(&counts2[e], 1);
    int slot = base[e] + pos;
    row_token[slot] = i >> 1;
    row_w[slot] = r_w[i];
}

// ---------------------------------------------------------------------------
// 6) residual copy: out = tokens
// ---------------------------------------------------------------------------
__global__ void moe_copy_kernel(const float4* __restrict__ in, float4* __restrict__ out,
                                int n4) {
    int i = blockIdx.x * 256 + threadIdx.x;
    if (i < n4) out[i] = in[i];
}

// ---------------------------------------------------------------------------
// 7) FFN: per 48-row tile of one expert:
//    h = gelu(x @ w1 + b1); y = h @ w2 + b2; out[token] += weight * y (atomic)
// LDS layout (overlaid regions, ~50.7KB total):
//   region0 [0, 25344):     fp32 async staging half-tile (24*256*4=24576B), later sh_h
//   region1 [25344, 50688): sh_x (bf16 x tile, TM*LDH*2)
// ---------------------------------------------------------------------------
__global__ __launch_bounds__(256) void moe_ffn_kernel(
    const float* __restrict__ tokens,
    const unsigned short* __restrict__ w1t, const float* __restrict__ b1,
    const unsigned short* __restrict__ w2t, const float* __restrict__ b2,
    const int* __restrict__ base, const int* __restrict__ row_token,
    const float* __restrict__ row_w, float* __restrict__ out) {

    __shared__ __align__(16) char sh_buf[TM * LDH * 2 * 2];  // 2 regions of TM*LDH ushorts
    __shared__ int   sh_tok[TM];
    __shared__ float sh_w[TM];

    unsigned short* sh_h = (unsigned short*)sh_buf;                  // region0 (after GEMM1)
    float*          sh_xf = (float*)sh_buf;                          // region0 (staging)
    unsigned short* sh_x = (unsigned short*)(sh_buf + TM * LDH * 2); // region1

    const int row0 = blockIdx.x * TM;
    if (row0 >= base[NEXP]) return;               // block-uniform exit
    int e = 0;
#pragma unroll
    for (int i = 1; i < NEXP; ++i) if (row0 >= base[i]) e = i;

    const int tid = threadIdx.x;
    if (tid < TM) {
        int t = row_token[row0 + tid];
        float w = row_w[row0 + tid];
        if (t < 0) { t = 0; w = 0.0f; }           // padded slot contributes nothing
        sh_tok[tid] = t;
        sh_w[tid] = w;
    }
    __syncthreads();

#if HAS_ASYNC_LDS
    // Async global->LDS gather of x rows (fp32), two half-tiles through region0,
    // converting each half to bf16 into sh_x. Tracked by ASYNCcnt.
    for (int hs = 0; hs < 2; ++hs) {
        for (int idx = tid; idx < TMH * 64; idx += 256) {
            int rl = idx >> 6, c4 = idx & 63;
            const float* gsrc = tokens + (size_t)sh_tok[hs * TMH + rl] * D_DIM + c4 * 4;
            float* ldst = sh_xf + rl * D_DIM + c4 * 4;
            __builtin_amdgcn_global_load_async_to_lds_b128(
                (glb_v4i_ptr)(glb_void*)gsrc, (lds_v4i_ptr)(lds_void*)ldst, 0, 0);
        }
        wait_async0();
        __syncthreads();
        for (int idx = tid; idx < TMH * 64; idx += 256) {
            int rl = idx >> 6, c4 = idx & 63;
            float4 v = *(const float4*)(sh_xf + rl * D_DIM + c4 * 4);
            usvec4 hv;
            hv[0] = f32_to_bf16(v.x); hv[1] = f32_to_bf16(v.y);
            hv[2] = f32_to_bf16(v.z); hv[3] = f32_to_bf16(v.w);
            *(usvec4*)(sh_x + (hs * TMH + rl) * LDH + c4 * 4) = hv;
        }
        __syncthreads();
    }
#else
    // Fallback: direct global->VGPR->convert->LDS staging
    for (int idx = tid; idx < TM * 64; idx += 256) {
        int r = idx >> 6, c4 = idx & 63;
        float4 v = *(const float4*)(tokens + (size_t)sh_tok[r] * D_DIM + c4 * 4);
        usvec4 hv;
        hv[0] = f32_to_bf16(v.x); hv[1] = f32_to_bf16(v.y);
        hv[2] = f32_to_bf16(v.z); hv[3] = f32_to_bf16(v.w);
        *(usvec4*)(sh_x + r * LDH + c4 * 4) = hv;
    }
    __syncthreads();
#endif

    const int wv = tid >> 5, lane = tid & 31;
    const unsigned short* w1e = w1t + ((size_t)e << 16);
    const unsigned short* w2e = w2t + ((size_t)e << 16);

    // ---- GEMM1: h = gelu(x @ w1 + b1) ----
#pragma unroll
    for (int ct = 0; ct < 2; ++ct) {
        const int ncb = wv * 32 + ct * 16;
        v8f acc[3] = {v8f{}, v8f{}, v8f{}};
#pragma unroll
        for (int k = 0; k < 8; ++k) {
            const int k0 = k * 32;
            FragBF fb = load_frag_b(w1e, ncb, k0, lane);
            if (k < 7)   // prefetch next k-step B fragment (global_prefetch_b8)
                __builtin_prefetch(w1e + (ncb + (lane & 15)) * D_DIM + k0 + 32, 0, 1);
#pragma unroll
            for (int m = 0; m < 3; ++m) {
                FragBF fa = load_frag_a(sh_x, m * 16, k0, lane);
                acc[m] = wmma_bf16(fa, fb, acc[m]);
            }
        }
        const int n = ncb + (lane & 15);
        const float bias = b1[e * D_DIM + n];
#pragma unroll
        for (int m = 0; m < 3; ++m)
#pragma unroll
            for (int v = 0; v < 8; ++v) {
                float x = acc[m][v] + bias;
                x = 0.5f * x * (1.0f + erff(x * 0.70710678118654752f));  // exact gelu
                int row = m * 16 + (lane >> 4) * 8 + v;
                sh_h[row * LDH + n] = f32_to_bf16(x);
            }
    }
    __syncthreads();

    // ---- GEMM2: y = h @ w2 + b2; scatter out[token] += w*y ----
#pragma unroll
    for (int ct = 0; ct < 2; ++ct) {
        const int ncb = wv * 32 + ct * 16;
        v8f acc[3] = {v8f{}, v8f{}, v8f{}};
#pragma unroll
        for (int k = 0; k < 8; ++k) {
            const int k0 = k * 32;
            FragBF fb = load_frag_b(w2e, ncb, k0, lane);
            if (k < 7)
                __builtin_prefetch(w2e + (ncb + (lane & 15)) * D_DIM + k0 + 32, 0, 1);
#pragma unroll
            for (int m = 0; m < 3; ++m) {
                FragBF fa = load_frag_a(sh_h, m * 16, k0, lane);
                acc[m] = wmma_bf16(fa, fb, acc[m]);
            }
        }
        const int n = ncb + (lane & 15);
        const float bias = b2[e * D_DIM + n];
#pragma unroll
        for (int m = 0; m < 3; ++m)
#pragma unroll
            for (int v = 0; v < 8; ++v) {
                int row = m * 16 + (lane >> 4) * 8 + v;
                float y = acc[m][v] + bias;
                unsafeAtomicAdd(&out[(size_t)sh_tok[row] * D_DIM + n], sh_w[row] * y);
            }
    }
}

// ---------------------------------------------------------------------------
// host launch
// ---------------------------------------------------------------------------
extern "C" void kernel_launch(void* const* d_in, const int* in_sizes, int n_in,
                              void* d_out, int out_size, void* d_ws, size_t ws_size,
                              hipStream_t stream) {
    const float* tokens   = (const float*)d_in[0];
    const float* router_w = (const float*)d_in[1];
    const float* w1       = (const float*)d_in[2];
    const float* b1       = (const float*)d_in[3];
    const float* w2       = (const float*)d_in[4];
    const float* b2       = (const float*)d_in[5];
    float* out = (float*)d_out;

    const int N  = in_sizes[0] / D_DIM;     // 65536 tokens
    const int N2 = 2 * N;                   // routed rows
    const int MROW = N2 + NEXP * TM + 256;  // padded row capacity

    char* ws = (char*)d_ws;
    size_t off = 0;
    auto take = [&](size_t bytes) { void* p = ws + off; off = (off + bytes + 255) & ~(size_t)255; return p; };
    int*   counts    = (int*)take(NEXP * 4);
    int*   counts2   = (int*)take(NEXP * 4);
    int*   base      = (int*)take((NEXP + 1) * 4);
    int*   r_e       = (int*)take((size_t)N2 * 4);
    float* r_w       = (float*)take((size_t)N2 * 4);
    int*   row_token = (int*)take((size_t)MROW * 4);
    float* row_w     = (float*)take((size_t)MROW * 4);
    unsigned short* w1t = (unsigned short*)take((size_t)NEXP * D_DIM * D_DIM * 2);
    unsigned short* w2t = (unsigned short*)take((size_t)NEXP * D_DIM * D_DIM * 2);
    (void)ws_size; (void)n_in; (void)out_size;

    // 1) init
    moe_init_kernel<<<(MROW + 255) / 256, 256, 0, stream>>>(counts, counts2, row_token, row_w, MROW);
    // 2) router
    moe_router_kernel<<<(N + 7) / 8, 256, 0, stream>>>(tokens, router_w, r_e, r_w, counts, N);
    // 3) weight transpose+convert
    moe_prep_kernel<<<(NEXP * D_DIM * D_DIM) / 256, 256, 0, stream>>>(w1, w2, w1t, w2t);
    // 4) scan
    moe_scan_kernel<<<1, 64, 0, stream>>>(counts, base);
    // 5) gather
    moe_gather_kernel<<<(N2 + 255) / 256, 256, 0, stream>>>(r_e, r_w, base, counts2, row_token, row_w, N2);
    // 6) residual copy
    moe_copy_kernel<<<(N * D_DIM / 4 + 255) / 256, 256, 0, stream>>>((const float4*)tokens, (float4*)out, N * D_DIM / 4);
    // 7) FFN (fixed worst-case grid; blocks past base[8] exit)
    const int tiles = (N2 + TM - 1) / TM + NEXP;
    moe_ffn_kernel<<<tiles, 256, 0, stream>>>(tokens, w1t, b1, w2t, b2, base, row_token, row_w, out);
}
